// LinearAttention_70677981823074
// MI455X (gfx1250) — compile-verified, matrix operand order checked by reading, not run on hardware
//
#include <hip/hip_runtime.h>
#include <hip/hip_bf16.h>

typedef __bf16 bf16;
typedef __attribute__((ext_vector_type(16))) __bf16 v16bf;
typedef __attribute__((ext_vector_type(8)))  __bf16 bf16x8;
typedef __attribute__((ext_vector_type(8)))  float  v8f;

#define REAL 4097     // dim == h == 4097
#define NCTX 4096     // N (decay mask extent)
#define PAD  4224     // 33 * 128, multiple of BM/BN/BK
#define LAMBDA 0.9f

// ---------------------------------------------------------------------------
// CDNA5 async global->LDS copy of 16 bytes (per lane), tracked by ASYNCcnt.
// vdst = LDS byte offset (low 32 bits of the generic shared pointer),
// vaddr = 64-bit global address, GV mode (saddr = off).
// ---------------------------------------------------------------------------
__device__ __forceinline__ void async_ld16(bf16* lds, const bf16* g) {
    unsigned ldsoff = (unsigned)(unsigned long long)lds;
    asm volatile("global_load_async_to_lds_b128 %0, %1, off"
                 :: "v"(ldsoff), "v"(g) : "memory");
}
// ASYNCcnt completes in issue order: waiting <=N with N copies just issued
// guarantees everything issued before them has landed.
__device__ __forceinline__ void wait_async0() {
    asm volatile("s_wait_asynccnt 0x0" ::: "memory");
}
__device__ __forceinline__ void wait_async2() {
    asm volatile("s_wait_asynccnt 0x2" ::: "memory");
}
__device__ __forceinline__ void wait_async4() {
    asm volatile("s_wait_asynccnt 0x4" ::: "memory");
}

// ---------------------------------------------------------------------------
// fp32 (REAL x REAL, ld=REAL) -> bf16 (PAD x PAD, ld=PAD), zero-padded.
// ---------------------------------------------------------------------------
__global__ __launch_bounds__(256) void cvt_pad_kernel(const float* __restrict__ src,
                                                      bf16* __restrict__ dst) {
    int c = blockIdx.x * 256 + threadIdx.x;
    int r = blockIdx.y;
    if (c >= PAD) return;
    float v = (r < REAL && c < REAL) ? src[(size_t)r * REAL + c] : 0.0f;
    dst[(size_t)r * PAD + c] = (bf16)v;
}

// ---------------------------------------------------------------------------
// Transposed convert: dst[c, r] = bf16(src[r, c]) with zero padding.
// 64x64 tile through LDS -> coalesced reads AND b128-coalesced writes.
// ---------------------------------------------------------------------------
__global__ __launch_bounds__(256) void cvt_padT_kernel(const float* __restrict__ src,
                                                       bf16* __restrict__ dst) {
    __shared__ bf16 tile[64][72];   // 144B row stride (16B aligned)
    const int t  = threadIdx.x;
    const int r0 = blockIdx.y * 64;
    const int c0 = blockIdx.x * 64;

    int rr = t >> 2;
    int cs = (t & 3) * 16;
#pragma unroll
    for (int j = 0; j < 16; ++j) {
        int cc = cs + j;
        float v = (r0 + rr < REAL && c0 + cc < REAL)
                      ? src[(size_t)(r0 + rr) * REAL + (c0 + cc)] : 0.0f;
        tile[cc][rr] = (bf16)v;
    }
    __syncthreads();

    int cc2 = t >> 2;
    int rs  = (t & 3) * 16;
    bf16x8 v0 = *reinterpret_cast<const bf16x8*>(&tile[cc2][rs]);
    bf16x8 v1 = *reinterpret_cast<const bf16x8*>(&tile[cc2][rs + 8]);
    bf16* d = dst + (size_t)(c0 + cc2) * PAD + r0 + rs;
    *reinterpret_cast<bf16x8*>(d)     = v0;
    *reinterpret_cast<bf16x8*>(d + 8) = v1;
}

// ---------------------------------------------------------------------------
// Decay-mask GEMM replaced by a suffix scan (rows = mask index c, cols = i):
//   leftT[c, i] = T1t[c, i] + LAMBDA * leftT[c+1, i]
// ---------------------------------------------------------------------------
__global__ __launch_bounds__(256) void decay_scan_kernel(const bf16* __restrict__ t1t,
                                                         bf16* __restrict__ leftT) {
    int i = blockIdx.x * 256 + threadIdx.x;
    if (i >= PAD) return;
    for (int c = NCTX; c < PAD; ++c) leftT[(size_t)c * PAD + i] = (bf16)0.0f;
    float acc = 0.0f;
#pragma unroll 8
    for (int c = NCTX - 1; c >= 0; --c) {
        acc = (float)t1t[(size_t)c * PAD + i] + LAMBDA * acc;
        leftT[(size_t)c * PAD + i] = (bf16)acc;
    }
}

// ---------------------------------------------------------------------------
// Fragment load: operand tiles live in LDS as [row][k] (stride 40).
// ISA 7.12.2 16-bit layout: lane group kg gets K runs [kg*8, kg*8+8) and
// [16+kg*8, 16+kg*8+8)  ->  two ds_load_b128 per fragment.
// ---------------------------------------------------------------------------
__device__ __forceinline__ v16bf load_frag(const bf16 (*S)[40], int row, int kg) {
    bf16x8 lo = *reinterpret_cast<const bf16x8*>(&S[row][kg * 8]);
    bf16x8 hi = *reinterpret_cast<const bf16x8*>(&S[row][16 + kg * 8]);
    return __builtin_shufflevector(lo, hi, 0, 1, 2, 3, 4, 5, 6, 7,
                                   8, 9, 10, 11, 12, 13, 14, 15);
}

// ---------------------------------------------------------------------------
// bf16 GEMM, f32 accumulate. A logical (M x K); B ALWAYS stored transposed
// as Bt[n, k] (ld PAD). 128x128x32 block tile, 8 waves, 2x4 wmma per wave.
// Double-buffered LDS: next tile's async global->LDS copies fly underneath
// the current tile's WMMAs (wait <= just-issued count, in-order ASYNCcnt).
//   TRANSA=0: A stored [m,k]  -> async verbatim (4 async copies/lane total)
//   TRANSA=1: A stored [k,m]  -> VGPR transpose-scatter (B async: 2/lane)
//   CMODE 1: Ct[n, m] = bf16(acc)  (b128-packed column-run stores, ld PAD)
//   CMODE 2: out[m, n] = Z[m, n] + acc*alpha, fp32, guarded to REAL x REAL
// ---------------------------------------------------------------------------
template <bool TRANSA, int CMODE>
__global__ __launch_bounds__(256) void gemm_bf16_kernel(const bf16* __restrict__ A,
                                                        const bf16* __restrict__ Bt,
                                                        bf16* __restrict__ Ct,
                                                        float* __restrict__ Cf,
                                                        const float* __restrict__ Zadd,
                                                        float alpha) {
    __shared__ bf16 As[2][128][40];   // 80B row stride (16B aligned)
    __shared__ bf16 Bs[2][128][40];

    const int tid  = threadIdx.x;
    const int lane = tid & 31;
    const int wid  = tid >> 5;       // 0..7
    const int wm   = wid >> 1;       // 0..3 -> 32-row slab
    const int wn   = wid & 1;        // 0..1 -> 64-col slab
    const int m0   = blockIdx.y * 128;
    const int n0   = blockIdx.x * 128;
    const int kg   = lane >> 4;      // lane group (0/1)
    const int lr   = lane & 15;

    auto stageAB = [&](int buf, int kk) {
        if constexpr (!TRANSA) {
            int row = tid >> 1;
            int seg = (tid & 1) * 16;
            const bf16* g = A + (size_t)(m0 + row) * PAD + kk + seg;
            async_ld16(&As[buf][row][seg],     g);
            async_ld16(&As[buf][row][seg + 8], g + 8);
        } else {
            int rowk = tid >> 3;            // storage row = k
            int seg  = (tid & 7) * 16;      // m offset
            const bf16x8* gp =
                reinterpret_cast<const bf16x8*>(A + (size_t)(kk + rowk) * PAD + m0 + seg);
            bf16x8 v0 = gp[0];
            bf16x8 v1 = gp[1];
#pragma unroll
            for (int j = 0; j < 8; ++j) As[buf][seg + j][rowk]     = v0[j];
#pragma unroll
            for (int j = 0; j < 8; ++j) As[buf][seg + 8 + j][rowk] = v1[j];
        }
        {
            int row = tid >> 1;
            int seg = (tid & 1) * 16;
            const bf16* g = Bt + (size_t)(n0 + row) * PAD + kk + seg;
            async_ld16(&Bs[buf][row][seg],     g);
            async_ld16(&Bs[buf][row][seg + 8], g + 8);
        }
    };

    v8f acc[2][4] = {};

    stageAB(0, 0);                             // prologue: fill buffer 0

    int it = 0;
    for (int k0 = 0; k0 < PAD; k0 += 32, ++it) {
        const int cur = it & 1;
        const bool hasNext = (k0 + 32) < PAD;

        if (hasNext) stageAB(cur ^ 1, k0 + 32);   // prefetch next tile (async)

        // current tile guaranteed landed; next tile's copies stay in flight
        if (hasNext) {
            if constexpr (!TRANSA) wait_async4(); else wait_async2();
        } else {
            wait_async0();
        }
        __syncthreads();

        v16bf afrag[2];
        v16bf bfrag[4];
#pragma unroll
        for (int mi = 0; mi < 2; ++mi)
            afrag[mi] = load_frag(As[cur], wm * 32 + mi * 16 + lr, kg);
#pragma unroll
        for (int ni = 0; ni < 4; ++ni)
            bfrag[ni] = load_frag(Bs[cur], wn * 64 + ni * 16 + lr, kg);

#pragma unroll
        for (int mi = 0; mi < 2; ++mi)
#pragma unroll
            for (int ni = 0; ni < 4; ++ni)
                acc[mi][ni] = __builtin_amdgcn_wmma_f32_16x16x32_bf16(
                    false, afrag[mi], false, bfrag[ni], (short)0, acc[mi][ni],
                    false, false);

        __syncthreads();   // protect cur from being restaged next iteration
    }

    // ---- epilogue: C/D layout M = r + 8*kg, N = lr ----
#pragma unroll
    for (int mi = 0; mi < 2; ++mi) {
#pragma unroll
        for (int ni = 0; ni < 4; ++ni) {
            if constexpr (CMODE == 1) {
                bf16x8 pk;
#pragma unroll
                for (int r = 0; r < 8; ++r) pk[r] = (bf16)acc[mi][ni][r];
                int gmb = m0 + wm * 32 + mi * 16 + kg * 8;     // contiguous 8-run
                int gn  = n0 + wn * 64 + ni * 16 + lr;
                *reinterpret_cast<bf16x8*>(Ct + (size_t)gn * PAD + gmb) = pk;
            } else {
#pragma unroll
                for (int r = 0; r < 8; ++r) {
                    int gm = m0 + wm * 32 + mi * 16 + r + 8 * kg;
                    int gn = n0 + wn * 64 + ni * 16 + lr;
                    if (gm < REAL && gn < REAL)
                        Cf[(size_t)gm * REAL + gn] =
                            Zadd[(size_t)gm * REAL + gn] + acc[mi][ni][r] * alpha;
                }
            }
        }
    }
}

// ---------------------------------------------------------------------------
// Orchestration (B operands always stored transposed [n][k]):
//   Zt   = Z^T (bf16, padded)     -- B for GEMM1/2 AND A for GEMM3
//   T1t  = (P @ Z)^T              gemm<false,1>(Pb, Zt)
//   T2t  = (Q @ Z)^T              gemm<false,1>(Qb, Zt)
//   leftT= suffix-scan(T1t)       (replaces the @M GEMM)
//   Rt   = (Z^T @ T2)^T           gemm<false,1>(Zt, T2t)
//   out  = Z + (left @ R)/N       gemm<true,2>(leftT, Rt)
// ---------------------------------------------------------------------------
extern "C" void kernel_launch(void* const* d_in, const int* in_sizes, int n_in,
                              void* d_out, int out_size, void* d_ws, size_t ws_size,
                              hipStream_t stream) {
    (void)in_sizes; (void)n_in; (void)out_size; (void)ws_size;
    const float* Z = (const float*)d_in[0];
    const float* P = (const float*)d_in[1];
    const float* Q = (const float*)d_in[2];
    float* out = (float*)d_out;

    const size_t MAT = (size_t)PAD * PAD;       // elements per padded matrix
    bf16* Zt  = (bf16*)d_ws;
    bf16* Pb  = Zt  + MAT;
    bf16* Qb  = Pb  + MAT;
    bf16* T1t = Qb  + MAT;
    bf16* T2t = T1t + MAT;
    bf16* Rt  = Pb;      // P dead after GEMM-1
    bf16* LFT = Qb;      // Q dead after GEMM-2

    dim3 cgrid((PAD + 255) / 256, PAD);
    cvt_pad_kernel<<<cgrid, 256, 0, stream>>>(P, Pb);
    cvt_pad_kernel<<<cgrid, 256, 0, stream>>>(Q, Qb);
    dim3 tgrid(PAD / 64, PAD / 64);
    cvt_padT_kernel<<<tgrid, 256, 0, stream>>>(Z, Zt);

    dim3 ggrid(PAD / 128, PAD / 128);
    gemm_bf16_kernel<false, 1><<<ggrid, 256, 0, stream>>>(Pb, Zt, T1t, nullptr, nullptr, 1.0f);
    gemm_bf16_kernel<false, 1><<<ggrid, 256, 0, stream>>>(Qb, Zt, T2t, nullptr, nullptr, 1.0f);
    decay_scan_kernel<<<(PAD + 255) / 256, 256, 0, stream>>>(T1t, LFT);
    gemm_bf16_kernel<false, 1><<<ggrid, 256, 0, stream>>>(Zt, T2t, Rt, nullptr, nullptr, 1.0f);
    gemm_bf16_kernel<true, 2><<<ggrid, 256, 0, stream>>>(LFT, Rt, nullptr, out, Z,
                                                         1.0f / (float)NCTX);
}
